// InstantNGPFieldRepresentation_73083163509187
// MI455X (gfx1250) — matrix operand order repaired and checked by reading, not compiled, not run on hardware
//
#include <hip/hip_runtime.h>
#include <math.h>

typedef __attribute__((ext_vector_type(16))) _Float16 v16h;
typedef __attribute__((ext_vector_type(8)))  float    v8f;

#define NLV     16
#define LOG2T   19
#define TSZ     (1u << LOG2T)
#define TMASK   (TSZ - 1u)
#define HID     128
#define DENC    32
#define EPSN    1e-4f
#define NWAVE   4      // waves (16-pt tiles) per block

// floor(16 * 1.5^l) for l = 0..15
__constant__ float kRes[16] = {
    16.f, 24.f, 36.f, 54.f, 81.f, 121.f, 182.f, 273.f,
    410.f, 615.f, 922.f, 1383.f, 2075.f, 3113.f, 4670.f, 7006.f};

// Fused Instant-NGP fwd + coord-VJP kernel.
// Block = 128 threads = 4 waves; each wave owns a 16-point tile.
// Weights live in LDS in BOTH row-major and transposed layouts so every
// WMMA B-fragment is one contiguous, 32B-aligned v16h load (2x ds_load_b128).
__global__ __launch_bounds__(128, 1)
void ngp_fused_kernel(const float* __restrict__ coords,
                      const float* __restrict__ tables,
                      const float* __restrict__ W1,   // [32][128]
                      const float* __restrict__ W2,   // [128][128]
                      const float* __restrict__ W3,   // [128][2]
                      float* __restrict__ out,        // [2N complex floats][3N normals]
                      int N) {
  // weights: natural (row-major) + transposed copies
  __shared__ __attribute__((aligned(32))) _Float16 W2s[HID * HID];   // W2[k][m]
  __shared__ __attribute__((aligned(32))) _Float16 W2t[HID * HID];   // W2t[m][k]
  __shared__ __attribute__((aligned(32))) _Float16 W1s[DENC * HID];  // W1[e][m]
  __shared__ __attribute__((aligned(32))) _Float16 W1t[HID * DENC];  // W1t[m][e]
  __shared__ float W3s[HID * 2];
  // per-wave tiles
  __shared__ __attribute__((aligned(32))) _Float16 xt [NWAVE][16 * DENC];
  __shared__ __attribute__((aligned(32))) _Float16 p1 [NWAVE][16 * HID];  // pre1 -> dpre1
  __shared__ __attribute__((aligned(32))) _Float16 p2 [NWAVE][16 * HID];  // pre2 -> dpre2
  __shared__ __attribute__((aligned(32))) _Float16 dxs[NWAVE][16 * DENC];
  __shared__ float outv[NWAVE][16 * 2];
  __shared__ float dch [NWAVE][16 * 3];

  const int tid  = threadIdx.x;
  const int wv   = tid >> 5;
  const int lane = tid & 31;
  const int hi   = lane >> 4;   // half-wave select
  const int lp   = lane & 15;   // local point AND WMMA N-column

  // ---- stage weights (f16, both layouts) ----------------------------------
  for (int i = tid; i < HID * HID; i += 128) {
    const int r = i >> 7, c = i & 127;
    const _Float16 v = (_Float16)W2[i];
    W2s[i] = v;
    W2t[c * HID + r] = v;
  }
  for (int i = tid; i < DENC * HID; i += 128) {
    const int r = i >> 7, c = i & 127;   // r = enc dim, c = hidden
    const _Float16 v = (_Float16)W1[i];
    W1s[i] = v;
    W1t[c * DENC + r] = v;
  }
  for (int i = tid; i < HID * 2; i += 128) W3s[i] = W3[i];
  __syncthreads();

  int gp = blockIdx.x * (16 * NWAVE) + wv * 16 + lp;
  if (gp >= N) gp = N - 1;  // clamp (select): EXEC stays full for WMMA

  const float cx = coords[3 * gp + 0];
  const float cy = coords[3 * gp + 1];
  const float cz = coords[3 * gp + 2];

  // ---- hash-grid encode: this lane does 8 levels of its point -------------
  const int lbase = hi * 8;
  float gacc[8][6];  // d(feat_f)/d(w_d), registers
  const float2* tab2 = (const float2*)tables;
  #pragma unroll
  for (int il = 0; il < 8; ++il) {
    const int l = lbase + il;
    const float res = kRes[l];
    const float px = cx * res, py = cy * res, pz = cz * res;
    const float fx = floorf(px), fy = floorf(py), fz = floorf(pz);
    const float wx = px - fx, wy = py - fy, wz = pz - fz;
    const unsigned ix0 = (unsigned)(int)fx;
    const unsigned iy0 = (unsigned)(int)fy;
    const unsigned iz0 = (unsigned)(int)fz;
    float f0 = 0.f, f1 = 0.f;
    float g0 = 0.f, g1 = 0.f, g2 = 0.f, g3 = 0.f, g4 = 0.f, g5 = 0.f;
    #pragma unroll
    for (int c = 0; c < 8; ++c) {
      const int bx = c & 1, by = (c >> 1) & 1, bz = (c >> 2) & 1;
      const unsigned h = ((ix0 + (unsigned)bx) ^
                         ((iy0 + (unsigned)by) * 2654435761u) ^
                         ((iz0 + (unsigned)bz) * 805459861u)) & TMASK;
      const float2 t = tab2[(size_t)l * TSZ + h];   // L2-resident gather
      const float mx = bx ? wx : 1.f - wx;
      const float my = by ? wy : 1.f - wy;
      const float mz = bz ? wz : 1.f - wz;
      const float w   = mx * my * mz;
      const float dwx = (bx ? 1.f : -1.f) * my * mz;
      const float dwy = (by ? 1.f : -1.f) * mx * mz;
      const float dwz = (bz ? 1.f : -1.f) * mx * my;
      f0 += w * t.x;   f1 += w * t.y;
      g0 += dwx * t.x; g1 += dwx * t.y;
      g2 += dwy * t.x; g3 += dwy * t.y;
      g4 += dwz * t.x; g5 += dwz * t.y;
    }
    gacc[il][0] = g0; gacc[il][1] = g1;
    gacc[il][2] = g2; gacc[il][3] = g3;
    gacc[il][4] = g4; gacc[il][5] = g5;
    xt[wv][lp * DENC + 2 * l + 0] = (_Float16)f0;
    xt[wv][lp * DENC + 2 * l + 1] = (_Float16)f1;
  }
  __syncthreads();

  // K mapping used here for both A and B (contiguous, not the ISA table order):
  //   lanes 0-15: K = 0..15; lanes 16-31: K = 16..31 (element s -> K = 16*hi + s)
  const int kb   = hi * 8;    // unused table-order K base
  const int koff = hi * 16;   // B half-wave K base

  // ---- layer 1: pre1 = x @ W1 (16x32 * 32x128), B from W1t (contig) -------
  v16h a1;
  #pragma unroll
  for (int i = 0; i < 16; ++i) {
    const int K = koff + i;
    a1[i] = xt[wv][lp * DENC + K];
  }
  #pragma unroll
  for (int n = 0; n < 8; ++n) {
    const v16h b = *(const v16h*)&W1t[(n * 16 + lp) * DENC + koff];
    v8f c = {};
    c = __builtin_amdgcn_wmma_f32_16x16x32_f16(false, a1, false, b,
                                               (short)0, c, false, false);
    #pragma unroll
    for (int r = 0; r < 8; ++r)
      p1[wv][(r + 8 * hi) * HID + n * 16 + lp] = (_Float16)c[r];
  }

  // ---- layer 2: pre2 = sin(pre1) @ W2, B from W2t (contig) ----------------
  v16h a2[4];
  #pragma unroll
  for (int kc = 0; kc < 4; ++kc)
    #pragma unroll
    for (int i = 0; i < 16; ++i) {
      const int K = kc * 32 + koff + i;
      a2[kc][i] = (_Float16)__sinf((float)p1[wv][lp * HID + K]);
    }
  #pragma unroll
  for (int n = 0; n < 8; ++n) {
    v8f c = {};
    #pragma unroll
    for (int kc = 0; kc < 4; ++kc) {
      const v16h b = *(const v16h*)&W2t[(n * 16 + lp) * HID + kc * 32 + koff];
      c = __builtin_amdgcn_wmma_f32_16x16x32_f16(false, a2[kc], false, b,
                                                 (short)0, c, false, false);
    }
    #pragma unroll
    for (int r = 0; r < 8; ++r)
      p2[wv][(r + 8 * hi) * HID + n * 16 + lp] = (_Float16)c[r];
  }

  // ---- layer 3 + complex output: out_j = sin(pre2) . W3[:,j] --------------
  {
    float acc = 0.f;
    const int j = hi;
    for (int k = 0; k < HID; ++k)
      acc += __sinf((float)p2[wv][lp * HID + k]) * W3s[k * 2 + j];
    outv[wv][lp * 2 + j] = acc;
    out[2 * (size_t)gp + j] = acc;   // complex64 interleaved (re, im)
  }

  // ---- backward: dpre2 = (sign(out) @ W3^T) * cos(pre2), in place ---------
  {
    const float o0 = outv[wv][lp * 2 + 0];
    const float o1 = outv[wv][lp * 2 + 1];
    const float s0 = (o0 > 0.f) ? 1.f : ((o0 < 0.f) ? -1.f : 0.f);
    const float s1 = (o1 > 0.f) ? 1.f : ((o1 < 0.f) ? -1.f : 0.f);
    const int k0 = hi * 64;
    for (int k = k0; k < k0 + 64; ++k) {
      const float g  = s0 * W3s[k * 2 + 0] + s1 * W3s[k * 2 + 1];
      const float pv = (float)p2[wv][lp * HID + k];
      p2[wv][lp * HID + k] = (_Float16)(g * __cosf(pv));
    }
  }

  // ---- dpre1 = (dpre2 @ W2^T) * cos(pre1), B from W2s natural (contig) ----
  v16h a3[4];
  #pragma unroll
  for (int kc = 0; kc < 4; ++kc)
    #pragma unroll
    for (int i = 0; i < 16; ++i) {
      const int K = kc * 32 + koff + i;
      a3[kc][i] = p2[wv][lp * HID + K];
    }
  #pragma unroll
  for (int n = 0; n < 8; ++n) {
    v8f c = {};
    #pragma unroll
    for (int kc = 0; kc < 4; ++kc) {
      const v16h b = *(const v16h*)&W2s[(n * 16 + lp) * HID + kc * 32 + koff];
      c = __builtin_amdgcn_wmma_f32_16x16x32_f16(false, a3[kc], false, b,
                                                 (short)0, c, false, false);
    }
    #pragma unroll
    for (int r = 0; r < 8; ++r) {
      const int M = r + 8 * hi, col = n * 16 + lp;
      const float pre = (float)p1[wv][M * HID + col];
      p1[wv][M * HID + col] = (_Float16)(c[r] * __cosf(pre));
    }
  }

  // ---- dx = dpre1 @ W1^T (16x128 * 128x32), B from W1s natural (contig) ---
  v16h a4[4];
  #pragma unroll
  for (int kc = 0; kc < 4; ++kc)
    #pragma unroll
    for (int i = 0; i < 16; ++i) {
      const int K = kc * 32 + koff + i;
      a4[kc][i] = p1[wv][lp * HID + K];
    }
  #pragma unroll
  for (int n = 0; n < 2; ++n) {
    v8f c = {};
    #pragma unroll
    for (int kc = 0; kc < 4; ++kc) {
      const v16h b = *(const v16h*)&W1s[(n * 16 + lp) * HID + kc * 32 + koff];
      c = __builtin_amdgcn_wmma_f32_16x16x32_f16(false, a4[kc], false, b,
                                                 (short)0, c, false, false);
    }
    #pragma unroll
    for (int r = 0; r < 8; ++r)
      dxs[wv][(r + 8 * hi) * DENC + n * 16 + lp] = (_Float16)c[r];
  }

  // ---- dcoord via chain rule through trilinear weights --------------------
  float pdx = 0.f, pdy = 0.f, pdz = 0.f;
  #pragma unroll
  for (int il = 0; il < 8; ++il) {
    const int l = lbase + il;
    const float r  = kRes[l];
    const float d0 = (float)dxs[wv][lp * DENC + 2 * l + 0];
    const float d1 = (float)dxs[wv][lp * DENC + 2 * l + 1];
    pdx += r * (d0 * gacc[il][0] + d1 * gacc[il][1]);
    pdy += r * (d0 * gacc[il][2] + d1 * gacc[il][3]);
    pdz += r * (d0 * gacc[il][4] + d1 * gacc[il][5]);
  }
  if (hi) {
    dch[wv][lp * 3 + 0] = pdx;
    dch[wv][lp * 3 + 1] = pdy;
    dch[wv][lp * 3 + 2] = pdz;
  }
  __syncthreads();
  if (!hi) {
    float nx = -(pdx + dch[wv][lp * 3 + 0]);
    float ny = -(pdy + dch[wv][lp * 3 + 1]);
    float nz = -(pdz + dch[wv][lp * 3 + 2]);
    const float nn  = nx * nx + ny * ny + nz * nz;
    const float inv = 1.f / sqrtf(fmaxf(nn, EPSN));
    nx *= inv; ny *= inv; nz *= inv;
    if (nx != nx) nx = 0.f;
    if (ny != ny) ny = 0.f;
    if (nz != nz) nz = 0.f;
    const size_t base = (size_t)2 * (size_t)N + (size_t)3 * (size_t)gp;
    out[base + 0] = nx;
    out[base + 1] = ny;
    out[base + 2] = nz;
  }
}

extern "C" void kernel_launch(void* const* d_in, const int* in_sizes, int n_in,
                              void* d_out, int out_size, void* d_ws, size_t ws_size,
                              hipStream_t stream) {
  const float* coords = (const float*)d_in[0];
  const float* tables = (const float*)d_in[1];
  const float* W1     = (const float*)d_in[2];
  const float* W2     = (const float*)d_in[3];
  const float* W3     = (const float*)d_in[4];
  float* out = (float*)d_out;
  const int N = in_sizes[0] / 3;
  const int blocks = (N + 16 * NWAVE - 1) / (16 * NWAVE);  // 64 pts per block
  hipLaunchKernelGGL(ngp_fused_kernel, dim3(blocks), dim3(128), 0, stream,
                     coords, tables, W1, W2, W3, out, N);
}
